// LSTMLayer_26714696581323
// MI455X (gfx1250) — compile-verified, matrix operand order checked by reading, not run on hardware
//
#include <hip/hip_runtime.h>
#include <hip/hip_bf16.h>

// CDNA5 / gfx1250 LSTM layer.
// Per-step fused GEMM: gates[64 x 4096] = [x_t | h] (K=2048) x [W_ih; W_hh]^T
// on v_wmma_f32_16x16x32_bf16 with f32 accumulation. Weights pre-packed into
// WMMA-B fragment-native bf16 layout (contiguous 32B per lane per fragment).
// A operand staged through LDS: x-phase converts f32->bf16 in VALU, h-phase
// uses GLOBAL_LOAD_ASYNC_TO_LDS_B128 (ASYNCcnt) for a zero-VGPR copy.

typedef __attribute__((ext_vector_type(16))) __bf16 v16bf;
typedef __attribute__((ext_vector_type(8)))  float  v8f;

#define T_STEPS 512
#define BATCH   64
#define IDIM    1024
#define HDIM    1024
#define KTOT    2048          // I + H concatenated along K
#define NTOT    4096          // 4*H gate columns
#define KSTEPS  64            // KTOT / 32 (WMMA K = 32)
#define FRAG_ELEMS 512        // 32 lanes * 16 bf16 per B fragment

// ---------------------------------------------------------------------------
// Pack [W_ih; W_hh] (f32, each [4096 x 1024] row-major, row = gate col n)
// into bf16 WMMA-B fragment layout:
//   fragment f = nTile*KSTEPS + kStep, element e = lane*16 + j
//   n = nTile*16 + (lane & 15)
//   k = kStep*32 + (lane >> 4)*16 + j     (k < 1024 -> W_ih, else W_hh)
// ---------------------------------------------------------------------------
__global__ __launch_bounds__(256) void pack_weights(
    const float* __restrict__ Wih, const float* __restrict__ Whh,
    __bf16* __restrict__ Wpk)
{
    long tid   = (long)blockIdx.x * blockDim.x + threadIdx.x;  // 8M total
    long f     = tid >> 9;
    int  e     = (int)(tid & 511);
    int  nTile = (int)(f / KSTEPS);
    int  kStep = (int)(f % KSTEPS);
    int  lane  = e >> 4;
    int  j     = e & 15;
    int  n     = nTile * 16 + (lane & 15);
    int  k     = kStep * 32 + (lane >> 4) * 16 + j;
    float v = (k < IDIM) ? Wih[(long)n * IDIM + k]
                         : Whh[(long)n * HDIM + (k - IDIM)];
    Wpk[tid] = (__bf16)v;
}

__global__ __launch_bounds__(256) void init_state(
    const float* __restrict__ h0, const float* __restrict__ c0,
    __bf16* __restrict__ hbf, float* __restrict__ cws)
{
    int i = blockIdx.x * blockDim.x + threadIdx.x;
    if (i < BATCH * HDIM) {
        hbf[i] = (__bf16)h0[i];
        cws[i] = c0[i];
    }
}

// ---------------------------------------------------------------------------
// One LSTM timestep. Grid = 16 WGs (each owns 64 hidden columns across all 4
// gates), block = 256 threads = 8 wave32s. Wave w -> gate (w>>1), column half
// (w&1); each wave accumulates 4 (M) x 2 (N) = 8 WMMA tiles over 64 K-steps.
// ---------------------------------------------------------------------------
__global__ __launch_bounds__(256) void lstm_step(
    const float*  __restrict__ x_t,    // [B][I] f32 slice for this t
    const __bf16* __restrict__ h_in,   // [B][H] bf16
    const __bf16* __restrict__ Wpk,    // packed fragments
    const float*  __restrict__ b_ih,   // [4H]
    const float*  __restrict__ b_hh,   // [4H]
    float*        __restrict__ c_ws,   // [B][H] cell state (in/out)
    __bf16*       __restrict__ h_out,  // [B][H] bf16 for next step
    float*        __restrict__ out_t)  // [B][H] -> outputs[t]
{
    // 64KB LDS: first 4KB aliased as the A-staging buffer during the K loop,
    // whole buffer reused as the f32 gate-exchange buffer afterwards.
    __shared__ __align__(128) char smem[64 * 1024];
    __bf16* stageA = (__bf16*)smem;   // [64 rows][32 k] bf16
    float*  gbuf   = (float*)smem;    // [4 gates][64 rows][64 cols] f32

    const int tid   = threadIdx.x;
    const int lane  = tid & 31;
    const int wave  = tid >> 5;
    const int g     = wave >> 1;      // gate 0..3 (i,f,g,o)
    const int nhalf = wave & 1;       // which 32-column half
    const int j0    = blockIdx.x * 64;

    v8f acc[4][2];
    {
        v8f z = {};
        for (int mt = 0; mt < 4; ++mt)
            for (int nt = 0; nt < 2; ++nt) acc[mt][nt] = z;
    }

    const int row = tid >> 2;   // 0..63 (batch row for staging)
    const int seg = tid & 3;    // 8-element (16B) segment
    const int nTileBase = g * 64 + (j0 >> 4) + nhalf * 2;

    __bf16* stDst = stageA + row * 32 + seg * 8;
    // LDS byte address for async-to-LDS: low 32 bits of the generic address
    // (LDS aperture maps by truncation, ISA 10.2).
    const unsigned ldsoff = (unsigned)(uintptr_t)stDst;

    auto mma_step = [&](int ks) {
        v16bf bfrag[2];
#pragma unroll
        for (int nt = 0; nt < 2; ++nt) {
            long fidx = (long)(nTileBase + nt) * KSTEPS + ks;
            const __bf16* bp = Wpk + fidx * FRAG_ELEMS + lane * 16;
            bfrag[nt] = *reinterpret_cast<const v16bf*>(bp);
            __builtin_prefetch(bp + FRAG_ELEMS, 0, 3);   // global_prefetch_b8
        }
#pragma unroll
        for (int mt = 0; mt < 4; ++mt) {
            const __bf16* ap =
                stageA + (mt * 16 + (lane & 15)) * 32 + (lane >> 4) * 16;
            v16bf afrag = *reinterpret_cast<const v16bf*>(ap);
#pragma unroll
            for (int nt = 0; nt < 2; ++nt) {
                acc[mt][nt] = __builtin_amdgcn_wmma_f32_16x16x32_bf16(
                    false, afrag, false, bfrag[nt],
                    (short)0, acc[mt][nt], false, false);
            }
        }
    };

    // ---- Phase 1: K = 0..1023 from x_t (f32 -> bf16 convert staging) ----
    for (int ks = 0; ks < 32; ++ks) {
        const float4* s4 = reinterpret_cast<const float4*>(
            x_t + (long)row * IDIM + ks * 32 + seg * 8);
        float4 f0 = s4[0];
        float4 f1 = s4[1];
        stDst[0] = (__bf16)f0.x; stDst[1] = (__bf16)f0.y;
        stDst[2] = (__bf16)f0.z; stDst[3] = (__bf16)f0.w;
        stDst[4] = (__bf16)f1.x; stDst[5] = (__bf16)f1.y;
        stDst[6] = (__bf16)f1.z; stDst[7] = (__bf16)f1.w;
        __syncthreads();
        mma_step(ks);
        __syncthreads();
    }

    // ---- Phase 2: K = 1024..2047 from h (async copy to LDS, ASYNCcnt) ----
    for (int ks = 32; ks < 64; ++ks) {
        const __bf16* gsrc = h_in + (long)row * HDIM + (ks - 32) * 32 + seg * 8;
        asm volatile("global_load_async_to_lds_b128 %0, %1, off"
                     :: "v"(ldsoff), "v"(gsrc) : "memory");
        asm volatile("s_wait_asynccnt 0" ::: "memory");
        __syncthreads();
        mma_step(ks);
        __syncthreads();
    }

    // ---- scatter accumulators to the gate-exchange buffer ----
    // C/D layout: VGPR r, lane l -> m = (l>>4)*8 + r, n = l & 15
#pragma unroll
    for (int mt = 0; mt < 4; ++mt)
#pragma unroll
        for (int nt = 0; nt < 2; ++nt)
#pragma unroll
            for (int r = 0; r < 8; ++r) {
                int m  = mt * 16 + (lane >> 4) * 8 + r;
                int nc = nhalf * 32 + nt * 16 + (lane & 15);
                gbuf[(g * 64 + m) * 64 + nc] = acc[mt][nt][r];
            }
    __syncthreads();

    // ---- elementwise LSTM cell (torch gate order i,f,g,o) ----
    const int b  = tid >> 2;
    const int jb = (tid & 3) * 16;
    for (int i = 0; i < 16; ++i) {
        int j   = jb + i;
        int col = j0 + j;
        float iv = gbuf[(0 * 64 + b) * 64 + j] + b_ih[col]            + b_hh[col];
        float fv = gbuf[(1 * 64 + b) * 64 + j] + b_ih[HDIM + col]     + b_hh[HDIM + col];
        float gv = gbuf[(2 * 64 + b) * 64 + j] + b_ih[2 * HDIM + col] + b_hh[2 * HDIM + col];
        float ov = gbuf[(3 * 64 + b) * 64 + j] + b_ih[3 * HDIM + col] + b_hh[3 * HDIM + col];
        iv = 1.0f / (1.0f + __expf(-iv));
        fv = 1.0f / (1.0f + __expf(-fv));
        gv = tanhf(gv);
        ov = 1.0f / (1.0f + __expf(-ov));
        long idx   = (long)b * HDIM + col;
        float cn   = fv * c_ws[idx] + iv * gv;
        float hn   = ov * tanhf(cn);
        c_ws[idx]  = cn;
        out_t[idx] = hn;
        h_out[idx] = (__bf16)hn;
    }
}

__global__ __launch_bounds__(256) void finalize(
    const float* __restrict__ outs, const float* __restrict__ cws,
    float* __restrict__ dout)
{
    int i = blockIdx.x * blockDim.x + threadIdx.x;
    if (i < BATCH * HDIM) {
        const long base = (long)T_STEPS * BATCH * HDIM;
        dout[base + i]                = outs[(long)(T_STEPS - 1) * BATCH * HDIM + i];
        dout[base + BATCH * HDIM + i] = cws[i];
    }
}

static inline size_t align256(size_t x) { return (x + 255) & ~(size_t)255; }

extern "C" void kernel_launch(void* const* d_in, const int* in_sizes, int n_in,
                              void* d_out, int out_size, void* d_ws, size_t ws_size,
                              hipStream_t stream) {
    const float* x   = (const float*)d_in[0];   // [T][B][I]
    const float* h0  = (const float*)d_in[1];   // [B][H]
    const float* c0  = (const float*)d_in[2];   // [B][H]
    const float* Wih = (const float*)d_in[3];   // [4H][I]
    const float* Whh = (const float*)d_in[4];   // [4H][H]
    const float* bih = (const float*)d_in[5];   // [4H]
    const float* bhh = (const float*)d_in[6];   // [4H]
    float* out = (float*)d_out;                 // outputs | h_fin | c_fin

    char*   ws   = (char*)d_ws;
    size_t  off  = 0;
    __bf16* Wpk  = (__bf16*)(ws + off); off += align256((size_t)NTOT * KTOT * sizeof(__bf16));
    __bf16* hbf0 = (__bf16*)(ws + off); off += align256((size_t)BATCH * HDIM * sizeof(__bf16));
    __bf16* hbf1 = (__bf16*)(ws + off); off += align256((size_t)BATCH * HDIM * sizeof(__bf16));
    float*  cws  = (float*) (ws + off); off += align256((size_t)BATCH * HDIM * sizeof(float));
    (void)ws_size; (void)in_sizes; (void)n_in; (void)out_size;

    // 1) pack weights into WMMA fragment layout (8M fragment elements)
    pack_weights<<<(NTOT * KTOT) / 256, 256, 0, stream>>>(Wih, Whh, Wpk);
    // 2) initial state
    init_state<<<(BATCH * HDIM + 255) / 256, 256, 0, stream>>>(h0, c0, hbf0, cws);
    // 3) sequential recurrence: one kernel per timestep, ping-pong h buffers
    for (int t = 0; t < T_STEPS; ++t) {
        const __bf16* hin  = (t & 1) ? hbf1 : hbf0;
        __bf16*       hout = (t & 1) ? hbf0 : hbf1;
        lstm_step<<<HDIM / 64, 256, 0, stream>>>(
            x + (long)t * BATCH * IDIM, hin, Wpk, bih, bhh, cws, hout,
            out + (long)t * BATCH * HDIM);
    }
    // 4) h_fin / c_fin
    finalize<<<(BATCH * HDIM + 255) / 256, 256, 0, stream>>>(out, cws, out);
}
